// RingPredictor_1829656068678
// MI455X (gfx1250) — matrix operand order repaired and checked by reading, not compiled, not run on hardware
//
#include <hip/hip_runtime.h>
#include <hip/hip_bf16.h>

#define F_INC 64
#define F_E   16
#define HEADS 4
#define CH    64
#define HC    256

typedef __attribute__((ext_vector_type(16))) _Float16 v16h;
typedef __attribute__((ext_vector_type(8)))  _Float16 v8h;
typedef __attribute__((ext_vector_type(8)))  float    v8f;

__device__ __forceinline__ v16h cat16(v8h a, v8h b) {
    return __builtin_shufflevector(a, b, 0, 1, 2, 3, 4, 5, 6, 7,
                                   8, 9, 10, 11, 12, 13, 14, 15);
}

// ---------- order-preserving float <-> uint for atomicMax ----------
__device__ __forceinline__ unsigned f2ord(float f) {
    unsigned u = __float_as_uint(f);
    return (u & 0x80000000u) ? ~u : (u | 0x80000000u);
}
__device__ __forceinline__ float ord2f(unsigned u) {
    return __uint_as_float((u & 0x80000000u) ? (u ^ 0x80000000u) : ~u);
}
__device__ __forceinline__ float lrelu(float x) { return x > 0.f ? x : 0.2f * x; }

// ---------- utility ----------
__global__ void zero_f32(float* __restrict__ p, int cnt) {
    int t = blockIdx.x * blockDim.x + threadIdx.x;
    if (t < cnt) p[t] = 0.f;
}

// ---------- self-loop attr: deg + sum of incoming edge attrs ----------
__global__ void edge_deg(const float* __restrict__ eattr, const int* __restrict__ dst,
                         float* __restrict__ deg, float* __restrict__ loop, int e) {
    int t = blockIdx.x * blockDim.x + threadIdx.x;
    if (t >= e * F_E) return;
    int ed = t >> 4, f = t & 15;
    int d = dst[ed];
    atomicAdd(&loop[d * F_E + f], eattr[t]);
    if (f == 0) atomicAdd(&deg[d], 1.0f);
}
__global__ void loop_fin(float* __restrict__ loop, const float* __restrict__ deg, int n) {
    int t = blockIdx.x * blockDim.x + threadIdx.x;
    if (t >= n * F_E) return;
    float dg = deg[t >> 4];
    loop[t] /= (dg < 1.f ? 1.f : dg);
}

// ---------- v[f,h] = sum_c We[f, h*64+c] * atte[h*64+c] ----------
__global__ void compute_v(const float* __restrict__ We, const float* __restrict__ atte,
                          float* __restrict__ v) {
    int t = threadIdx.x;
    if (t >= F_E * HEADS) return;
    int f = t >> 2, h = t & 3;
    float s = 0.f;
    for (int c = 0; c < CH; ++c) s += We[f * HC + h * CH + c] * atte[h * CH + c];
    v[f * HEADS + h] = s;
}

// ---------- WMMA GEMM: C[M,256] = A[M,K] @ W[K,256], f16 in / f32 acc ----------
// Block: 256 thr = 8 waves, tile 128(M) x 64(N); wave tile 32x32 -> 4 WMMA/K-step.
__global__ __launch_bounds__(256) void gemm_wmma_256(
    const float* __restrict__ A, const float* __restrict__ W,
    float* __restrict__ C, int M, int K)
{
    __shared__ _Float16 As[128][40];  // row stride 80B (16B multiple)
    __shared__ _Float16 Bt[64][40];   // B transposed: [n][k]
    const int tid  = threadIdx.x;
    const int lane = tid & 31;
    const int wave = tid >> 5;
    const int bm0  = blockIdx.x * 128;
    const int bn0  = blockIdx.y * 64;
    const int mt   = (wave & 3) * 32;   // wave M offset in block tile
    const int nb   = (wave >> 2) * 32;  // wave N offset in block tile
    const int ml   = lane & 15;
    const int klo  = (lane >= 16) ? 8 : 0;   // A lane-half K offset
    const int kb   = (lane >= 16) ? 16 : 0;  // unused; B loads use klo
    v8f acc00 = {}, acc01 = {}, acc10 = {}, acc11 = {};

    for (int k0 = 0; k0 < K; k0 += 32) {
        { // stage A 128x32: thread -> row tid/2, cols (tid&1)*16..+15, one guard
            int r = tid >> 1, c0 = (tid & 1) * 16;
            int gr = bm0 + r;
            float4 f0 = {}, f1 = {}, f2 = {}, f3 = {};
            if (gr < M) {
                const float4* p = (const float4*)(A + (size_t)gr * K + k0 + c0);
                f0 = p[0]; f1 = p[1]; f2 = p[2]; f3 = p[3];
            }
            v8h h0 = { (_Float16)f0.x, (_Float16)f0.y, (_Float16)f0.z, (_Float16)f0.w,
                       (_Float16)f1.x, (_Float16)f1.y, (_Float16)f1.z, (_Float16)f1.w };
            v8h h1 = { (_Float16)f2.x, (_Float16)f2.y, (_Float16)f2.z, (_Float16)f2.w,
                       (_Float16)f3.x, (_Float16)f3.y, (_Float16)f3.z, (_Float16)f3.w };
            *(v8h*)&As[r][c0]     = h0;
            *(v8h*)&As[r][c0 + 8] = h1;
        }
        { // stage B transposed: thread -> col tid&63, k (tid>>6)*8..+7
            int c = tid & 63, kq = (tid >> 6) * 8;
            v8h hb;
            #pragma unroll
            for (int i = 0; i < 8; ++i)
                hb[i] = (_Float16)W[(size_t)(k0 + kq + i) * HC + bn0 + c];
            *(v8h*)&Bt[c][kq] = hb;
        }
        __syncthreads();
        // fragments: contiguous per-lane chunks -> ds_load_b128 pairs
        v16h af0 = cat16(*(const v8h*)&As[mt + ml][klo],
                         *(const v8h*)&As[mt + ml][16 + klo]);
        v16h af1 = cat16(*(const v8h*)&As[mt + 16 + ml][klo],
                         *(const v8h*)&As[mt + 16 + ml][16 + klo]);
        v16h bf0 = cat16(*(const v8h*)&Bt[nb + ml][klo],
                         *(const v8h*)&Bt[nb + ml][16 + klo]);
        v16h bf1 = cat16(*(const v8h*)&Bt[nb + 16 + ml][klo],
                         *(const v8h*)&Bt[nb + 16 + ml][16 + klo]);
        acc00 = __builtin_amdgcn_wmma_f32_16x16x32_f16(false, af0, false, bf0,
                                                       (short)0, acc00, false, false);
        acc01 = __builtin_amdgcn_wmma_f32_16x16x32_f16(false, af0, false, bf1,
                                                       (short)0, acc01, false, false);
        acc10 = __builtin_amdgcn_wmma_f32_16x16x32_f16(false, af1, false, bf0,
                                                       (short)0, acc10, false, false);
        acc11 = __builtin_amdgcn_wmma_f32_16x16x32_f16(false, af1, false, bf1,
                                                       (short)0, acc11, false, false);
        __syncthreads();
    }
    // C/D layout: VGPR r -> M=r (lanes 0-15) / M=8+r (lanes 16-31); N = lane%16
    const int mo = (lane >= 16) ? 8 : 0;
    #pragma unroll
    for (int r = 0; r < 8; ++r) {
        int r0 = bm0 + mt + r + mo;
        int r1 = r0 + 16;
        int cA = bn0 + nb + ml;
        if (r0 < M) {
            C[(size_t)r0 * HC + cA]      = acc00[r];
            C[(size_t)r0 * HC + cA + 16] = acc01[r];
        }
        if (r1 < M) {
            C[(size_t)r1 * HC + cA]      = acc10[r];
            C[(size_t)r1 * HC + cA + 16] = acc11[r];
        }
    }
}

// ---------- per-node attention dots: a_s, a_d (wave per node) ----------
__global__ __launch_bounds__(256) void node_att(
    const float* __restrict__ xs, const float* __restrict__ atts,
    const float* __restrict__ attd, float* __restrict__ a_s,
    float* __restrict__ a_d, int n)
{
    int wave = threadIdx.x >> 5, lane = threadIdx.x & 31;
    int node = blockIdx.x * 8 + wave;
    if (node >= n) return;
    float ss = 0.f, sd = 0.f;
    int c0 = lane * 8;
    #pragma unroll
    for (int i = 0; i < 8; ++i) {
        float v = xs[(size_t)node * HC + c0 + i];
        ss += v * atts[c0 + i];
        sd += v * attd[c0 + i];
    }
    #pragma unroll
    for (int o = 1; o < 8; o <<= 1) {  // reduce within 8-lane head groups
        ss += __shfl_xor(ss, o, 32);
        sd += __shfl_xor(sd, o, 32);
    }
    if ((lane & 7) == 0) {
        int h = lane >> 3;
        a_s[node * HEADS + h] = ss;
        a_d[node * HEADS + h] = sd;
    }
}

// ---------- self-loop logits + init running max ----------
__global__ void node_self(const float* __restrict__ loop, const float* __restrict__ v,
                          const float* __restrict__ a_s, const float* __restrict__ a_d,
                          float* __restrict__ lself, unsigned* __restrict__ m_ord, int n)
{
    int t = blockIdx.x * blockDim.x + threadIdx.x;
    if (t >= n * HEADS) return;
    int node = t >> 2, h = t & 3;
    float ae = 0.f;
    #pragma unroll
    for (int f = 0; f < F_E; ++f) ae += loop[node * F_E + f] * v[f * HEADS + h];
    float l = lrelu(a_s[t] + a_d[t] + ae);
    lself[t] = l;
    m_ord[t] = f2ord(l);
}

// ---------- per-edge logits + segment max ----------
__global__ void edge_logit(const float* __restrict__ eattr, const int* __restrict__ src,
                           const int* __restrict__ dst, const float* __restrict__ v,
                           const float* __restrict__ a_s, const float* __restrict__ a_d,
                           float* __restrict__ ledge, unsigned* __restrict__ m_ord, int e)
{
    int t = blockIdx.x * blockDim.x + threadIdx.x;
    if (t >= e) return;
    int s = src[t], d = dst[t];
    float ea[F_E];
    #pragma unroll
    for (int f = 0; f < F_E; ++f) ea[f] = eattr[(size_t)t * F_E + f];
    #pragma unroll
    for (int h = 0; h < HEADS; ++h) {
        float ae = 0.f;
        #pragma unroll
        for (int f = 0; f < F_E; ++f) ae += ea[f] * v[f * HEADS + h];
        float l = lrelu(a_s[s * HEADS + h] + a_d[d * HEADS + h] + ae);
        ledge[(size_t)t * HEADS + h] = l;
        atomicMax(&m_ord[d * HEADS + h], f2ord(l));
    }
}

__global__ void m_convert(unsigned* __restrict__ m, int cnt) {
    int t = blockIdx.x * blockDim.x + threadIdx.x;
    if (t < cnt) { unsigned u = m[t]; ((float*)m)[t] = ord2f(u); }
}

__global__ void denom_init(const float* __restrict__ lself, const float* __restrict__ m,
                           float* __restrict__ denom, int cnt) {
    int t = blockIdx.x * blockDim.x + threadIdx.x;
    if (t < cnt) denom[t] = expf(lself[t] - m[t]);
}

__global__ void edge_denom(const float* __restrict__ ledge, const int* __restrict__ dst,
                           const float* __restrict__ m, float* __restrict__ denom, int e) {
    int t = blockIdx.x * blockDim.x + threadIdx.x;
    if (t >= e) return;
    int d = dst[t];
    #pragma unroll
    for (int h = 0; h < HEADS; ++h)
        atomicAdd(&denom[d * HEADS + h],
                  expf(ledge[(size_t)t * HEADS + h] - m[d * HEADS + h]));
}

// ---------- aggregation: self contribution + bias ----------
__global__ __launch_bounds__(256) void agg_init(
    const float* __restrict__ xs, const float* __restrict__ lself,
    const float* __restrict__ m, const float* __restrict__ denom,
    const float* __restrict__ b, float* __restrict__ out)
{
    int node = blockIdx.x, c = threadIdx.x, h = c >> 6;
    float a = expf(lself[node * HEADS + h] - m[node * HEADS + h]) /
              (denom[node * HEADS + h] + 1e-16f);
    out[(size_t)node * HC + c] = xs[(size_t)node * HC + c] * a + b[c];
}

// ---------- aggregation: one block per edge (L2-resident scatter) ----------
__global__ __launch_bounds__(256) void agg_edge(
    const float* __restrict__ xs, const int* __restrict__ src,
    const int* __restrict__ dst, const float* __restrict__ ledge,
    const float* __restrict__ m, const float* __restrict__ denom,
    float* __restrict__ out)
{
    int e = blockIdx.x;
    int s = src[e], d = dst[e];
    __shared__ float w[HEADS];
    if (threadIdx.x < HEADS) {
        int h = threadIdx.x;
        w[h] = expf(ledge[(size_t)e * HEADS + h] - m[d * HEADS + h]) /
               (denom[d * HEADS + h] + 1e-16f);
    }
    __syncthreads();
    int c = threadIdx.x;
    atomicAdd(&out[(size_t)d * HC + c], xs[(size_t)s * HC + c] * w[c >> 6]);
}

// ---------- BatchNorm (batch stats) + ReLU ----------
__global__ __launch_bounds__(256) void bn_stats(const float* __restrict__ x,
                                                float* __restrict__ sums, int n) {
    int c = threadIdx.x;
    float s = 0.f, s2 = 0.f;
    for (int r = blockIdx.x; r < n; r += gridDim.x) {
        float v = x[(size_t)r * HC + c];
        s += v; s2 += v * v;
    }
    atomicAdd(&sums[c], s);
    atomicAdd(&sums[HC + c], s2);
}
__global__ void bn_fin(const float* __restrict__ sums, float* __restrict__ musig, int n) {
    int c = threadIdx.x;
    float mu  = sums[c] / (float)n;
    float var = sums[HC + c] / (float)n - mu * mu;
    musig[c]      = mu;
    musig[HC + c] = rsqrtf(var + 1e-5f);
}
__global__ __launch_bounds__(256) void bn_apply(float* __restrict__ x,
                                                const float* __restrict__ musig,
                                                const float* __restrict__ g,
                                                const float* __restrict__ be) {
    int node = blockIdx.x, c = threadIdx.x;
    float v = x[(size_t)node * HC + c];
    v = g[c] * (v - musig[c]) * musig[HC + c] + be[c];
    x[(size_t)node * HC + c] = v > 0.f ? v : 0.f;
}

// ---------- output head: wave per node ----------
__global__ __launch_bounds__(256) void head_out(const float* __restrict__ h,
                                                const float* __restrict__ Wout,
                                                const float* __restrict__ bout,
                                                float* __restrict__ out, int n) {
    int wave = threadIdx.x >> 5, lane = threadIdx.x & 31;
    int node = blockIdx.x * 8 + wave;
    if (node >= n) return;
    float s = 0.f;
    #pragma unroll
    for (int i = 0; i < 8; ++i) {
        int c = lane * 8 + i;
        s += h[(size_t)node * HC + c] * Wout[c];
    }
    #pragma unroll
    for (int o = 16; o; o >>= 1) s += __shfl_xor(s, o, 32);
    if (lane == 0) out[node] = s + bout[0];
}

extern "C" void kernel_launch(void* const* d_in, const int* in_sizes, int n_in,
                              void* d_out, int out_size, void* d_ws, size_t ws_size,
                              hipStream_t stream) {
    const float* x     = (const float*)d_in[0];
    const int*   src   = (const int*)d_in[1];
    const int*   dst   = (const int*)d_in[2];
    const float* eattr = (const float*)d_in[3];
    const float* Wout  = (const float*)d_in[20];
    const float* bout  = (const float*)d_in[21];

    const int NN = in_sizes[0] / F_INC;   // 50000
    const int EE = in_sizes[1];           // 800000

    // workspace carve-up (floats)
    float* ws   = (float*)d_ws;
    float* xs    = ws;                 ws += (size_t)NN * HC;   // GEMM output
    float* agg   = ws;                 ws += (size_t)NN * HC;   // aggregation / hidden
    float* deg   = ws;                 ws += NN;
    float* loopA = ws;                 ws += (size_t)NN * F_E;
    float* vbuf  = ws;                 ws += 64;
    float* a_s   = ws;                 ws += (size_t)NN * HEADS;
    float* a_d   = ws;                 ws += (size_t)NN * HEADS;
    float* lself = ws;                 ws += (size_t)NN * HEADS;
    float* mbuf  = ws;                 ws += (size_t)NN * HEADS;
    float* denom = ws;                 ws += (size_t)NN * HEADS;
    float* ledge = ws;                 ws += (size_t)EE * HEADS;
    float* sums  = ws;                 ws += 2 * HC;
    float* musig = ws;                 ws += 2 * HC;

    // self-loop edge-attr (shared by both layers)
    zero_f32<<<(NN + 255) / 256, 256, 0, stream>>>(deg, NN);
    zero_f32<<<(NN * F_E + 255) / 256, 256, 0, stream>>>(loopA, NN * F_E);
    edge_deg<<<(EE * F_E + 255) / 256, 256, 0, stream>>>(eattr, dst, deg, loopA, EE);
    loop_fin<<<(NN * F_E + 255) / 256, 256, 0, stream>>>(loopA, deg, NN);

    auto layer = [&](const float* hin, int K,
                     const float* W, const float* atts, const float* attd,
                     const float* We, const float* atte, const float* b,
                     const float* g, const float* be) {
        compute_v<<<1, 64, 0, stream>>>(We, atte, vbuf);
        dim3 gg((NN + 127) / 128, HC / 64);
        gemm_wmma_256<<<gg, 256, 0, stream>>>(hin, W, xs, NN, K);
        node_att<<<(NN + 7) / 8, 256, 0, stream>>>(xs, atts, attd, a_s, a_d, NN);
        node_self<<<(NN * HEADS + 255) / 256, 256, 0, stream>>>(
            loopA, vbuf, a_s, a_d, lself, (unsigned*)mbuf, NN);
        edge_logit<<<(EE + 255) / 256, 256, 0, stream>>>(
            eattr, src, dst, vbuf, a_s, a_d, ledge, (unsigned*)mbuf, EE);
        m_convert<<<(NN * HEADS + 255) / 256, 256, 0, stream>>>((unsigned*)mbuf, NN * HEADS);
        denom_init<<<(NN * HEADS + 255) / 256, 256, 0, stream>>>(lself, mbuf, denom, NN * HEADS);
        edge_denom<<<(EE + 255) / 256, 256, 0, stream>>>(ledge, dst, mbuf, denom, EE);
        agg_init<<<NN, 256, 0, stream>>>(xs, lself, mbuf, denom, b, agg);
        agg_edge<<<EE, 256, 0, stream>>>(xs, src, dst, ledge, mbuf, denom, agg);
        zero_f32<<<2, 256, 0, stream>>>(sums, 2 * HC);
        bn_stats<<<256, 256, 0, stream>>>(agg, sums, NN);
        bn_fin<<<1, 256, 0, stream>>>(sums, musig, NN);
        bn_apply<<<NN, 256, 0, stream>>>(agg, musig, g, be);
    };

    layer(x,   F_INC, (const float*)d_in[4],  (const float*)d_in[5],
          (const float*)d_in[6],  (const float*)d_in[7],  (const float*)d_in[8],
          (const float*)d_in[9],  (const float*)d_in[10], (const float*)d_in[11]);
    layer(agg, HC,    (const float*)d_in[12], (const float*)d_in[13],
          (const float*)d_in[14], (const float*)d_in[15], (const float*)d_in[16],
          (const float*)d_in[17], (const float*)d_in[18], (const float*)d_in[19]);

    head_out<<<(NN + 7) / 8, 256, 0, stream>>>(agg, Wout, bout, (float*)d_out, NN);
}
